// Self_Attention_89979564851755
// MI455X (gfx1250) — matrix operand order repaired and checked by reading, not compiled, not run on hardware
//
#include <hip/hip_runtime.h>
#include <math.h>

// Problem constants (match reference: B=4, S=2048, D=512, fp32 in/out)
#define BB 4
#define SS 2048
#define DD 512

typedef _Float16 half_t;
typedef __attribute__((ext_vector_type(16))) _Float16 v16h;
typedef __attribute__((ext_vector_type(8)))  _Float16 v8h;
typedef __attribute__((ext_vector_type(8)))  float    v8f;
typedef unsigned int u32x4 __attribute__((ext_vector_type(4)));
typedef int          i32x4 __attribute__((ext_vector_type(4)));
typedef int          i32x8 __attribute__((ext_vector_type(8)));

#if defined(__has_builtin)
#if __has_builtin(__builtin_amdgcn_tensor_load_to_lds)
#define HAVE_TDM 1
#endif
#endif

union AFrag { v16h v; v8h h[2]; };

// A-matrix 16x32 f16 fragment (M x K), row-major source.
// Lane L<16: row M=L, K = kb+{0..7} and kb+{16..23}
// Lane L>=16: row M=L-16, K = kb+{8..15} and kb+{24..31}
__device__ __forceinline__ v16h load_a_frag(const half_t* row, int kb, int lane) {
    const int off = kb + ((lane & 16) ? 8 : 0);
    AFrag a;
    a.h[0] = *(const v8h*)(row + off);
    a.h[1] = *(const v8h*)(row + off + 16);
    return a.v;
}

// B-matrix 32x16 f16 fragment (K x N). `col` points at (k=0) of column N=lane&15,
// K-contiguous source (N-major / transposed). Lanes 0-15: K=kb+{0..7},{16..23}; 16-31: kb+{8..15},{24..31}.
__device__ __forceinline__ v16h load_b_frag(const half_t* col, int kb, int lane) {
    const int off = kb + ((lane & 16) ? 8 : 0);
    AFrag b;
    b.h[0] = *(const v8h*)(col + off);
    b.h[1] = *(const v8h*)(col + off + 16);
    return b.v;
}

__device__ __forceinline__ v8f wmma16(v16h a, v16h b, v8f c) {
    return __builtin_amdgcn_wmma_f32_16x16x32_f16(false, a, false, b, (short)0, c,
                                                  false, false);
}

// ---------------------------------------------------------------------------
// TDM: async 2D f16 tile load Global -> LDS (Tensor Data Mover, gfx1250)
// D# built per CDNA5 ISA 8.3/8.4: group0 = {count/type, lds_addr, global_addr},
// group1 = {data_size, tensor dims, tile dims, dim0 stride}; groups 2/3 zero (2D).
// ---------------------------------------------------------------------------
#ifdef HAVE_TDM
__device__ __forceinline__ void tdm_load_2d_f16(unsigned lds_off_bytes,
                                                const half_t* gptr,
                                                unsigned tensor_d0, unsigned tensor_d1,
                                                unsigned tile_d0, unsigned tile_d1,
                                                unsigned long long stride_elems) {
    const unsigned long long ga = (unsigned long long)(uintptr_t)gptr;
    u32x4 g0;
    g0[0] = 1u;                                                   // count=1, user D#
    g0[1] = lds_off_bytes;                                        // lds_addr
    g0[2] = (unsigned)ga;                                         // global_addr[31:0]
    g0[3] = (unsigned)((ga >> 32) & 0x01ffffffull) | (2u << 30);  // addr[56:32] | type=2
    i32x8 g1;
    g1[0] = 0x00010000;                                  // data_size=1 (2-byte), no pad/iter
    g1[1] = (int)((tensor_d0 & 0xffffu) << 16);          // [63:48]=tensor_dim0 lo16
    g1[2] = (int)((tensor_d0 >> 16) | ((tensor_d1 & 0xffffu) << 16));
    g1[3] = (int)((tensor_d1 >> 16) | (tile_d0 << 16));  // tile_dim0
    g1[4] = (int)(tile_d1 & 0xffffu);                    // tile_dim1 (tile_dim2=0)
    g1[5] = (int)(unsigned)stride_elems;                 // tensor_dim0_stride lo32
    g1[6] = (int)(unsigned)(stride_elems >> 32);         // stride hi16 (dim1 stride=0)
    g1[7] = 0;
    i32x4 z4 = {};
#if __has_include(<hip/amd_detail/amd_gfx1250_TDM.h>)
    i32x8 z8 = {};
    __builtin_amdgcn_tensor_load_to_lds(g0, g1, z4, z4, z8, 0);   // 6-arg toolchain
#else
    __builtin_amdgcn_tensor_load_to_lds(g0, g1, z4, z4, 0);       // 5-arg ROCm 7.2
#endif
}
#endif

// ---------------------------------------------------------------------------
// Kernel 1: convert X -> f16, and Wq/Wk/Wv -> f16 transposed (Wt[n][k] = W[k][n])
// ---------------------------------------------------------------------------
__global__ void convert_kernel(const float* __restrict__ X,
                               const float* __restrict__ Wq,
                               const float* __restrict__ Wk,
                               const float* __restrict__ Wv,
                               half_t* __restrict__ Xh,
                               half_t* __restrict__ Wqt,
                               half_t* __restrict__ Wkt,
                               half_t* __restrict__ Wvt) {
    const size_t NX = (size_t)BB * SS * DD;
    const size_t NW = (size_t)DD * DD;
    const size_t total = NX + 3 * NW;
    const size_t stride = (size_t)gridDim.x * blockDim.x;
    for (size_t i = (size_t)blockIdx.x * blockDim.x + threadIdx.x; i < total; i += stride) {
        if (i < NX) {
            Xh[i] = (half_t)X[i];
        } else {
            size_t j = i - NX;
            int w = (int)(j / NW);
            int r = (int)(j % NW);
            int n = r / DD;
            int k = r % DD;
            const float* W = (w == 0) ? Wq : (w == 1) ? Wk : Wv;
            half_t* Wt     = (w == 0) ? Wqt : (w == 1) ? Wkt : Wvt;
            Wt[(size_t)n * DD + k] = (half_t)W[(size_t)k * DD + n];
        }
    }
}

// ---------------------------------------------------------------------------
// Kernel 2: Q/K/V projections. 4 waves per block, 64x64 output panel per block.
// X panel (64x512) and W^T panel (64x512) staged in LDS by the TDM; each wave
// computes 16 rows x 64 cols (4 accumulators, A-fragment reused 4x).
// Q,K stored row-major f16; V stored transposed ([B][D][S]).
// ---------------------------------------------------------------------------
#define QKV_PANEL_BYTES (64 * DD * 2)   // 64 KB per panel

__global__ void qkv_kernel(const half_t* __restrict__ Xh,
                           const half_t* __restrict__ Wqt,
                           const half_t* __restrict__ Wkt,
                           const half_t* __restrict__ Wvt,
                           const float* __restrict__ bq,
                           const float* __restrict__ bk,
                           const float* __restrict__ bv,
                           half_t* __restrict__ Qh,
                           half_t* __restrict__ Kh,
                           half_t* __restrict__ Vt) {
    extern __shared__ char smem[];
    half_t* Xp = (half_t*)smem;                         // [64][DD]
    half_t* Wp = (half_t*)(smem + QKV_PANEL_BYTES);     // [64][DD]

    const int which = blockIdx.y;
    const int tn = blockIdx.x & 7;                      // 512/64 = 8 column panels
    const int tm = blockIdx.x >> 3;                     // 8192/64 = 128 row panels
    const int m0 = tm * 64;
    const int n0 = tn * 64;
    const int tid = threadIdx.x;
    const int lane = tid & 31;
    const int w = tid >> 5;

    const half_t* Wt   = (which == 0) ? Wqt : (which == 1) ? Wkt : Wvt;
    const float*  bias = (which == 0) ? bq  : (which == 1) ? bk  : bv;

#ifdef HAVE_TDM
    if (w == 0) {                                       // one wave issues both DMAs
        const unsigned lds0 = (unsigned)__builtin_amdgcn_groupstaticsize();
        tdm_load_2d_f16(lds0,                   Xh + (size_t)m0 * DD,
                        DD, (unsigned)(BB * SS), DD, 64, DD);
        tdm_load_2d_f16(lds0 + QKV_PANEL_BYTES, Wt + (size_t)n0 * DD,
                        DD, DD, DD, 64, DD);
        __builtin_amdgcn_s_wait_tensorcnt(0);
    }
#else
    {   // cooperative fallback copy (16B vectors)
        const uint4* gx = (const uint4*)(Xh + (size_t)m0 * DD);
        const uint4* gw = (const uint4*)(Wt + (size_t)n0 * DD);
        uint4* lx = (uint4*)Xp;
        uint4* lw = (uint4*)Wp;
        const int nvec = 64 * DD / 8;                   // 8 halves per uint4
        for (int i = tid; i < nvec; i += blockDim.x) { lx[i] = gx[i]; lw[i] = gw[i]; }
    }
#endif
    __syncthreads();

    const half_t* arow  = Xp + (size_t)(w * 16 + (lane & 15)) * DD;
    const half_t* bcol0 = Wp + (size_t)(lane & 15) * DD;

    v8f acc[4] = {};
    for (int k = 0; k < DD; k += 32) {
        // preload all fragments into independent registers, then issue WMMAs,
        // so the backend can overlap ds-loads with matrix ops (partial dscnt waits)
        v16h a  = load_a_frag(arow, k, lane);
        v16h b0 = load_b_frag(bcol0 + (size_t)(0 * 16) * DD, k, lane);
        v16h b1 = load_b_frag(bcol0 + (size_t)(1 * 16) * DD, k, lane);
        v16h b2 = load_b_frag(bcol0 + (size_t)(2 * 16) * DD, k, lane);
        v16h b3 = load_b_frag(bcol0 + (size_t)(3 * 16) * DD, k, lane);
        acc[0] = wmma16(a, b0, acc[0]);
        acc[1] = wmma16(a, b1, acc[1]);
        acc[2] = wmma16(a, b2, acc[2]);
        acc[3] = wmma16(a, b3, acc[3]);
    }

    const int mloc = w * 16 + ((lane >> 4) << 3);
    #pragma unroll
    for (int j = 0; j < 4; j++) {
        const int n = n0 + j * 16 + (lane & 15);
        const float bias_n = bias[n];
        #pragma unroll
        for (int r = 0; r < 8; r++) {
            const int m = m0 + mloc + r;
            const half_t val = (half_t)(acc[j][r] + bias_n);
            if (which == 0) {
                Qh[(size_t)m * DD + n] = val;
            } else if (which == 1) {
                Kh[(size_t)m * DD + n] = val;
            } else {
                const int bi = m / SS;
                const int s  = m % SS;
                Vt[((size_t)bi * DD + n) * SS + s] = val;
            }
        }
    }
}

// ---------------------------------------------------------------------------
// Kernel 3: attention, one workgroup (8 waves) per (batch, 16 query rows).
// Q slab (16x512) TDM-staged to LDS once. Phase 1: logits = Q*K^T -> LDS f32;
// waves own groups of 4 key tiles (A reused 4x). Phase 2: row softmax,
// exp -> f16 P in LDS. Phase 3: O = (P @ V)/rowsum; wave owns 64 D-columns.
// ---------------------------------------------------------------------------
#define LOGITS_BYTES (16 * SS * 4)
#define P_BYTES      (16 * SS * 2)
#define RED_BYTES    (16 * 16 * 4)
#define Q_OFF        (LOGITS_BYTES + P_BYTES + RED_BYTES + 128)
#define Q_BYTES      (16 * DD * 2)
#define ATTN_SMEM    (Q_OFF + Q_BYTES)

__global__ void attn_kernel(const half_t* __restrict__ Qh,
                            const half_t* __restrict__ Kh,
                            const half_t* __restrict__ Vt,
                            float* __restrict__ out) {
    extern __shared__ char smem[];
    float*  logits = (float*)smem;                                   // [16][SS]
    half_t* P      = (half_t*)(smem + LOGITS_BYTES);                 // [16][SS]
    float*  red    = (float*)(smem + LOGITS_BYTES + P_BYTES);        // [16][16]
    float*  rowmax = (float*)(smem + LOGITS_BYTES + P_BYTES + RED_BYTES);
    float*  rowsum = (float*)(smem + LOGITS_BYTES + P_BYTES + RED_BYTES + 64);
    half_t* Qp     = (half_t*)(smem + Q_OFF);                        // [16][DD]

    const int tid  = threadIdx.x;
    const int lane = tid & 31;
    const int w    = tid >> 5;                 // wave id 0..7
    const int bi   = blockIdx.x / (SS / 16);
    const int q0   = (blockIdx.x % (SS / 16)) * 16;

    // ---- Stage the 16x512 Q slab into LDS (TDM; shared by all 8 waves) ----
#ifdef HAVE_TDM
    if (w == 0) {
        const unsigned lds0 = (unsigned)__builtin_amdgcn_groupstaticsize();
        tdm_load_2d_f16(lds0 + Q_OFF, Qh + (size_t)(bi * SS + q0) * DD,
                        DD, (unsigned)(BB * SS), DD, 16, DD);
        __builtin_amdgcn_s_wait_tensorcnt(0);
    }
#else
    {
        const uint4* gq = (const uint4*)(Qh + (size_t)(bi * SS + q0) * DD);
        uint4* lq = (uint4*)Qp;
        const int nvec = 16 * DD / 8;          // 1024 x 16B
        for (int i = tid; i < nvec; i += blockDim.x) lq[i] = gq[i];
    }
#endif
    __syncthreads();

    // ---- Phase 1: logits = Q * K^T (unscaled, like the reference) ----
    const half_t* qrow = Qp + (size_t)(lane & 15) * DD;
    for (int g = 0; g < 4; g++) {
        const int kbase = (w + g * 8) * 64;    // group of 4 key tiles
        const half_t* kcol[4];
        #pragma unroll
        for (int j = 0; j < 4; j++)
            kcol[j] = Kh + (size_t)(bi * SS + kbase + j * 16 + (lane & 15)) * DD;
        v8f acc[4] = {};
        for (int d = 0; d < DD; d += 32) {
            v16h a  = load_a_frag(qrow, d, lane);
            v16h b0 = load_b_frag(kcol[0], d, lane);
            v16h b1 = load_b_frag(kcol[1], d, lane);
            v16h b2 = load_b_frag(kcol[2], d, lane);
            v16h b3 = load_b_frag(kcol[3], d, lane);
            acc[0] = wmma16(a, b0, acc[0]);
            acc[1] = wmma16(a, b1, acc[1]);
            acc[2] = wmma16(a, b2, acc[2]);
            acc[3] = wmma16(a, b3, acc[3]);
        }
        const int mb = (lane >> 4) << 3;
        #pragma unroll
        for (int j = 0; j < 4; j++) {
            const int n = kbase + j * 16 + (lane & 15);
            #pragma unroll
            for (int r = 0; r < 8; r++)
                logits[(size_t)(mb + r) * SS + n] = acc[j][r];
        }
    }
    __syncthreads();

    // ---- Phase 2: softmax over keys. 16 rows x 16 segments of 128 ----
    const int row = tid & 15;
    const int seg = tid >> 4;
    const int c0  = seg * (SS / 16);
    float m = -INFINITY;
    for (int j = 0; j < SS / 16; j++)
        m = fmaxf(m, logits[(size_t)row * SS + c0 + j]);
    red[row * 16 + seg] = m;
    __syncthreads();
    if (seg == 0) {
        float mm = red[row * 16];
        for (int j = 1; j < 16; j++) mm = fmaxf(mm, red[row * 16 + j]);
        rowmax[row] = mm;
    }
    __syncthreads();
    const float rm = rowmax[row];
    float s = 0.f;
    for (int j = 0; j < SS / 16; j++) {
        const float e = __expf(logits[(size_t)row * SS + c0 + j] - rm);
        P[(size_t)row * SS + c0 + j] = (half_t)e;   // unnormalized; divide at store
        s += e;
    }
    red[row * 16 + seg] = s;
    __syncthreads();
    if (seg == 0) {
        float ss = 0.f;
        for (int j = 0; j < 16; j++) ss += red[row * 16 + j];
        rowsum[row] = ss;
    }
    __syncthreads();

    // ---- Phase 3: O = (P @ V)/rowsum. Wave w owns D columns [w*64, w*64+64) ----
    const half_t* prow = P + (size_t)(lane & 15) * SS;
    const half_t* vcol[4];
    #pragma unroll
    for (int j = 0; j < 4; j++)
        vcol[j] = Vt + ((size_t)bi * DD + w * 64 + j * 16 + (lane & 15)) * SS;
    v8f acc[4] = {};
    for (int kk = 0; kk < SS; kk += 32) {
        v16h a  = load_a_frag(prow, kk, lane);
        v16h b0 = load_b_frag(vcol[0], kk, lane);
        v16h b1 = load_b_frag(vcol[1], kk, lane);
        v16h b2 = load_b_frag(vcol[2], kk, lane);
        v16h b3 = load_b_frag(vcol[3], kk, lane);
        acc[0] = wmma16(a, b0, acc[0]);
        acc[1] = wmma16(a, b1, acc[1]);
        acc[2] = wmma16(a, b2, acc[2]);
        acc[3] = wmma16(a, b3, acc[3]);
    }
    const int mb = (lane >> 4) << 3;
    #pragma unroll
    for (int j = 0; j < 4; j++) {
        const int n = w * 64 + j * 16 + (lane & 15);
        #pragma unroll
        for (int r = 0; r < 8; r++) {
            const int mrow = mb + r;
            out[(size_t)(bi * SS + q0 + mrow) * DD + n] = acc[j][r] / rowsum[mrow];
        }
    }
}

// ---------------------------------------------------------------------------
// Host-side launch
// ---------------------------------------------------------------------------
extern "C" void kernel_launch(void* const* d_in, const int* in_sizes, int n_in,
                              void* d_out, int out_size, void* d_ws, size_t ws_size,
                              hipStream_t stream) {
    (void)in_sizes; (void)n_in; (void)out_size; (void)ws_size;

    const float* X  = (const float*)d_in[0];
    const float* Wq = (const float*)d_in[1];
    const float* bq = (const float*)d_in[2];
    const float* Wk = (const float*)d_in[3];
    const float* bk = (const float*)d_in[4];
    const float* Wv = (const float*)d_in[5];
    const float* bv = (const float*)d_in[6];
    float* out = (float*)d_out;

    const size_t BSD = (size_t)BB * SS * DD;   // 4,194,304
    const size_t DxD = (size_t)DD * DD;        // 262,144

    half_t* Xh  = (half_t*)d_ws;
    half_t* Wqt = Xh + BSD;
    half_t* Wkt = Wqt + DxD;
    half_t* Wvt = Wkt + DxD;
    half_t* Qh  = Wvt + DxD;
    half_t* Kh  = Qh + BSD;
    half_t* Vt  = Kh + BSD;                    // total ~35 MB of workspace

    // 1) fp32 -> f16 conversion (+ weight transpose)
    {
        const size_t total = BSD + 3 * DxD;
        const int threads = 256;
        const int blocks = (int)((total + threads - 1) / threads);
        convert_kernel<<<blocks, threads, 0, stream>>>(X, Wq, Wk, Wv, Xh, Wqt, Wkt, Wvt);
    }

    // 2) Q/K/V projection GEMMs: 64x64 panels, TDM-staged operands in LDS
    {
        dim3 grid((BB * SS / 64) * (DD / 64), 3);   // 1024 x 3
        qkv_kernel<<<grid, 128, 2 * QKV_PANEL_BYTES, stream>>>(
            Xh, Wqt, Wkt, Wvt, bq, bk, bv, Qh, Kh, Vt);
    }

    // 3) attention (logits + softmax + PV) per 16-query slab
    {
        const int blocks = BB * (SS / 16);          // 512
        attn_kernel<<<blocks, 256, ATTN_SMEM, stream>>>(Qh, Kh, Vt, out);
    }
}
